// CrossAttentionModule_88596585382718
// MI455X (gfx1250) — matrix operand order repaired and checked by reading, not compiled, not run on hardware
//
#include <hip/hip_runtime.h>
#include <hip/hip_bf16.h>

// ---------------- problem constants ----------------
#define CDIM 2048
#define BDIM 8
#define NPIX 1024   // H*W = 32*32

typedef __attribute__((ext_vector_type(16))) __bf16 v16bf;
typedef __attribute__((ext_vector_type(8)))  float  v8f;

__device__ __forceinline__ unsigned short f2bf(float f) {
    union { float f; unsigned u; } c; c.f = f;
    unsigned u = c.u + 0x7FFFu + ((c.u >> 16) & 1u);
    return (unsigned short)(u >> 16);
}
__device__ __forceinline__ unsigned pack2(float a, float b) {
    return (unsigned)f2bf(a) | ((unsigned)f2bf(b) << 16);
}
__device__ __forceinline__ float wredSum(float v) {
    for (int o = 16; o > 0; o >>= 1) v += __shfl_xor(v, o, 32);
    return v;
}
__device__ __forceinline__ float wredMax(float v) {
    for (int o = 16; o > 0; o >>= 1) v = fmaxf(v, __shfl_xor(v, o, 32));
    return v;
}

// ---- CDNA5 async global->LDS copy (ASYNCcnt-tracked DMA into LDS) ----
__device__ __forceinline__ void async_copy16(unsigned ldsoff, const void* g) {
    unsigned long long ga = (unsigned long long)(uintptr_t)g;
    asm volatile("global_load_async_to_lds_b128 %0, %1, off"
                 :: "v"(ldsoff), "v"(ga) : "memory");
}
#if __has_builtin(__builtin_amdgcn_s_wait_asynccnt)
#define WAIT_ASYNC(n) __builtin_amdgcn_s_wait_asynccnt(n)
#else
#define WAIT_ASYNC(n) asm volatile("s_wait_asynccnt %0" :: "n"(n) : "memory")
#endif

// ---------------------------------------------------------------------------
// Kernel 1: per (b, 8-row c-chunk): xa[b,c]=sum_n x*a, xs[b,c]=sum_n x,
// and write transposed bf16 copy xT[b][n][c].
// grid = B * (C/8) = 2048 blocks, 256 threads.
// ---------------------------------------------------------------------------
__global__ void prep_kernel(const float* __restrict__ x, const float* __restrict__ att,
                            __bf16* __restrict__ xT, float* __restrict__ xa,
                            float* __restrict__ xs) {
    int b  = blockIdx.x >> 8;
    int c0 = (blockIdx.x & 255) << 3;
    int tid = threadIdx.x;
    int n0 = tid * 4;

    const float4 a4 = *(const float4*)(att + (size_t)b * NPIX + n0);

    float dotv[8], sumv[8];
    float xv[8][4];
#pragma unroll
    for (int i = 0; i < 8; ++i) {
        const float4 x4 = *(const float4*)(x + ((size_t)(b * CDIM + c0 + i)) * NPIX + n0);
        dotv[i] = x4.x * a4.x + x4.y * a4.y + x4.z * a4.z + x4.w * a4.w;
        sumv[i] = x4.x + x4.y + x4.z + x4.w;
        xv[i][0] = x4.x; xv[i][1] = x4.y; xv[i][2] = x4.z; xv[i][3] = x4.w;
    }
#pragma unroll
    for (int k = 0; k < 4; ++k) {
        uint4 pk;
        pk.x = pack2(xv[0][k], xv[1][k]);
        pk.y = pack2(xv[2][k], xv[3][k]);
        pk.z = pack2(xv[4][k], xv[5][k]);
        pk.w = pack2(xv[6][k], xv[7][k]);
        *(uint4*)(xT + ((size_t)b * NPIX + n0 + k) * CDIM + c0) = pk;
    }
#pragma unroll
    for (int i = 0; i < 8; ++i) { dotv[i] = wredSum(dotv[i]); sumv[i] = wredSum(sumv[i]); }

    __shared__ float rd[8][8], rs[8][8];
    int wave = tid >> 5, lane = tid & 31;
    if (lane == 0) {
#pragma unroll
        for (int i = 0; i < 8; ++i) { rd[wave][i] = dotv[i]; rs[wave][i] = sumv[i]; }
    }
    __syncthreads();
    if (tid < 8) {
        float s1 = 0.f, s2 = 0.f;
#pragma unroll
        for (int w = 0; w < 8; ++w) { s1 += rd[w][tid]; s2 += rs[w][tid]; }
        xa[b * CDIM + c0 + tid] = s1;
        xs[b * CDIM + c0 + tid] = s2;
    }
}

// sa[b] = sum_n attention[b,n]; grid = B, block = 256
__global__ void sa_kernel(const float* __restrict__ att, float* __restrict__ sa) {
    int b = blockIdx.x;
    float v = 0.f;
    for (int n = threadIdx.x; n < NPIX; n += 256) v += att[(size_t)b * NPIX + n];
    v = wredSum(v);
    __shared__ float r[8];
    int wave = threadIdx.x >> 5, lane = threadIdx.x & 31;
    if (lane == 0) r[wave] = v;
    __syncthreads();
    if (threadIdx.x == 0) {
        float s = 0.f;
        for (int w = 0; w < 8; ++w) s += r[w];
        sa[b] = s;
    }
}

// fp32 -> bf16 convert (for Wv); 4 elements/thread
__global__ void cvt_kernel(const float* __restrict__ W, __bf16* __restrict__ Wb, int ntot) {
    int idx = (blockIdx.x * 256 + threadIdx.x) * 4;
    if (idx < ntot) {
        const float4 w = *(const float4*)(W + idx);
        uint2 pk; pk.x = pack2(w.x, w.y); pk.y = pack2(w.z, w.w);
        *(uint2*)(Wb + idx) = pk;
    }
}

// ---------------------------------------------------------------------------
// Kernel 2: A[b,c] = Wq[c,:].xa[b,:] + bq[c]*sa[b]
//           S[b,c] = Wq[c,:].xs[b,:] + bq[c]*NPIX
// one wave per Wq row, all 8 batches accumulated in-register.
// ---------------------------------------------------------------------------
__global__ void qsum_kernel(const float* __restrict__ Wq, const float* __restrict__ bq,
                            const float* __restrict__ xa, const float* __restrict__ xs,
                            const float* __restrict__ sa, float* __restrict__ Ab,
                            float* __restrict__ Sb) {
    int wave = threadIdx.x >> 5, lane = threadIdx.x & 31;
    int c = blockIdx.x * 8 + wave;
    float accA[BDIM] = {}, accS[BDIM] = {};
    for (int e = lane * 4; e < CDIM; e += 128) {
        const float4 wq = *(const float4*)(Wq + (size_t)c * CDIM + e);
#pragma unroll
        for (int b = 0; b < BDIM; ++b) {
            const float4 va = *(const float4*)(xa + b * CDIM + e);
            const float4 vs = *(const float4*)(xs + b * CDIM + e);
            accA[b] += wq.x * va.x + wq.y * va.y + wq.z * va.z + wq.w * va.w;
            accS[b] += wq.x * vs.x + wq.y * vs.y + wq.z * vs.z + wq.w * vs.w;
        }
    }
#pragma unroll
    for (int b = 0; b < BDIM; ++b) { accA[b] = wredSum(accA[b]); accS[b] = wredSum(accS[b]); }
    if (lane == 0) {
        float bqc = bq[c];
#pragma unroll
        for (int b = 0; b < BDIM; ++b) {
            Ab[b * CDIM + c] = accA[b] + bqc * sa[b];
            Sb[b * CDIM + c] = accS[b] + bqc * (float)NPIX;
        }
    }
}

// ---------------------------------------------------------------------------
// Kernel 3: softmax rows. scores[d] = A*Wk[d] + S*bk[d]; P[b,c,d] bf16.
// one wave per (b,c) row; grid = B*C/8 = 2048 blocks, 256 threads.
// ---------------------------------------------------------------------------
__global__ void softmax_kernel(const float* __restrict__ Wk, const float* __restrict__ bk,
                               const float* __restrict__ Ab, const float* __restrict__ Sb,
                               __bf16* __restrict__ P) {
    int wave = threadIdx.x >> 5, lane = threadIdx.x & 31;
    int row = blockIdx.x * 8 + wave;   // row = b*C + c
    float Av = Ab[row], Sv = Sb[row];

    float sc[16][4];
    float mx = -3.4e38f;
#pragma unroll
    for (int ch = 0; ch < 16; ++ch) {
        int d = ch * 128 + lane * 4;
        const float4 wk = *(const float4*)(Wk + d);
        const float4 b4 = *(const float4*)(bk + d);
        sc[ch][0] = Av * wk.x + Sv * b4.x;
        sc[ch][1] = Av * wk.y + Sv * b4.y;
        sc[ch][2] = Av * wk.z + Sv * b4.z;
        sc[ch][3] = Av * wk.w + Sv * b4.w;
#pragma unroll
        for (int j = 0; j < 4; ++j) mx = fmaxf(mx, sc[ch][j]);
    }
    mx = wredMax(mx);
    float sum = 0.f;
#pragma unroll
    for (int ch = 0; ch < 16; ++ch)
#pragma unroll
        for (int j = 0; j < 4; ++j) { sc[ch][j] = __expf(sc[ch][j] - mx); sum += sc[ch][j]; }
    sum = wredSum(sum);
    float inv = __frcp_rn(sum);
    __bf16* prow = P + (size_t)row * CDIM;
#pragma unroll
    for (int ch = 0; ch < 16; ++ch) {
        int d = ch * 128 + lane * 4;
        uint2 pk;
        pk.x = pack2(sc[ch][0] * inv, sc[ch][1] * inv);
        pk.y = pack2(sc[ch][2] * inv, sc[ch][3] * inv);
        *(uint2*)(prow + d) = pk;
    }
}

// ---------------------------------------------------------------------------
// WMMA GEMM: D[M=2048][N=1024] = A[M][K=2048] (bf16 row-major) x Bt (bf16 [N][K]).
// Triple-buffered async global->LDS staging (128x32 tiles, 80B padded rows),
// ONE barrier per K-stage: stage st+3 reuses buffer st%3 and is issued only
// after barrier(st+1), which every wave reaches only after finishing
// consume(st) -> anti-overwrite ordering without a second barrier.
// block tile 128x128, 8 waves (4M x 2N), wave tile 32x64, K step 32.
// EPI==0: D = acc + bias[m], stored bf16 TRANSPOSED out[b][n][m]   (v = Wv@x + bv)
// EPI==1: out_f32[b][m][n] = gamma*acc + resid[b][m][n]            (final output)
// grid = B * 128 blocks, 256 threads.
// ---------------------------------------------------------------------------
union FragAB { uint4 u[2]; v16bf v; };

template <int EPI>
__global__ __launch_bounds__(256)
void gemm_wmma_kernel(const __bf16* __restrict__ A, size_t aStride,
                      const __bf16* __restrict__ Bt,
                      const float* __restrict__ bias,
                      const float* __restrict__ gammaPtr,
                      const float* __restrict__ resid,
                      __bf16* __restrict__ outBf,
                      float* __restrict__ outF) {
    constexpr int K = CDIM;
    constexpr int KSTEP = 32;
    constexpr int NSTAGE = K / KSTEP;      // 64
    constexpr int RS = 40;                 // padded row stride in elements (80B = 20 banks)
    constexpr int NBUF = 3;

    __shared__ __align__(16) __bf16 As[NBUF][128 * RS];
    __shared__ __align__(16) __bf16 Bs[NBUF][128 * RS];

    int b = blockIdx.x >> 7;
    int t = blockIdx.x & 127;
    int m_block = (t >> 3) * 128;
    int n_block = (t & 7) * 128;

    int tid = threadIdx.x;
    int wave = tid >> 5, lane = tid & 31;
    int wm = (wave & 3) * 32;              // wave M offset within block tile
    int wn = (wave >> 2) * 64;             // wave N offset within block tile
    int col = lane & 15, hi = lane >> 4;

    const __bf16* Ag = A + (size_t)b * aStride;
    const __bf16* Bg = Bt + (size_t)b * ((size_t)NPIX * (size_t)K);

    // tile-load segment assignment: 512 x 16B segments per tile, 2 per thread
    int s0r = tid >> 2,         s0q = tid & 3;
    int s1r = (tid + 256) >> 2, s1q = (tid + 256) & 3;

    unsigned aB[NBUF], bB[NBUF];
#pragma unroll
    for (int i = 0; i < NBUF; ++i) {
        aB[i] = (unsigned)(uintptr_t)&As[i][0];
        bB[i] = (unsigned)(uintptr_t)&Bs[i][0];
    }

    auto issue = [&](int stage, int buf) {
        int k0 = stage * KSTEP;
        unsigned ab = aB[buf], bb = bB[buf];
        async_copy16(ab + (unsigned)(s0r * 80 + s0q * 16),
                     Ag + (size_t)(m_block + s0r) * K + k0 + s0q * 8);
        async_copy16(ab + (unsigned)(s1r * 80 + s1q * 16),
                     Ag + (size_t)(m_block + s1r) * K + k0 + s1q * 8);
        async_copy16(bb + (unsigned)(s0r * 80 + s0q * 16),
                     Bg + (size_t)(n_block + s0r) * K + k0 + s0q * 8);
        async_copy16(bb + (unsigned)(s1r * 80 + s1q * 16),
                     Bg + (size_t)(n_block + s1r) * K + k0 + s1q * 8);
    };

    v8f acc[2][4] = {};

    issue(0, 0);
    int cbuf = 0, nbuf = 1;
    for (int st = 0; st < NSTAGE; ++st) {
        if (st + 1 < NSTAGE) {
            issue(st + 1, nbuf);
            WAIT_ASYNC(4);                 // stage st's 4 copies complete (per-wave, in order)
        } else {
            WAIT_ASYNC(0);
        }
        __syncthreads();                   // stage st tile visible to all waves

        FragAB af[2];
#pragma unroll
        for (int mi = 0; mi < 2; ++mi) {
            const __bf16* pa = &As[cbuf][(wm + mi * 16 + col) * RS + hi * 8];
            af[mi].u[0] = *(const uint4*)(pa);        // j=0..7  : K = hi*8 + j
            af[mi].u[1] = *(const uint4*)(pa + 16);   // j=8..15 : K = 16 + hi*8 + j-8
        }
        FragAB bfr[4];
#pragma unroll
        for (int ni = 0; ni < 4; ++ni) {
            const __bf16* pb = &Bs[cbuf][(wn + ni * 16 + col) * RS + hi * 8];
            bfr[ni].u[0] = *(const uint4*)(pb);       // K = hi*8 + 0..7
            bfr[ni].u[1] = *(const uint4*)(pb + 16);  // K = 16 + hi*8 + 0..7
        }
#pragma unroll
        for (int mi = 0; mi < 2; ++mi)
#pragma unroll
            for (int ni = 0; ni < 4; ++ni)
                acc[mi][ni] = __builtin_amdgcn_wmma_f32_16x16x32_bf16(
                    false, af[mi].v, false, bfr[ni].v, (short)0, acc[mi][ni], false, false);

        cbuf = (cbuf + 1 == NBUF) ? 0 : cbuf + 1;
        nbuf = (nbuf + 1 == NBUF) ? 0 : nbuf + 1;
    }

    if constexpr (EPI == 0) {
        // bias add + transposed bf16 store: out[b][n][m0..m0+7], one 16B store/lane/tile
#pragma unroll
        for (int mi = 0; mi < 2; ++mi) {
            int m0 = m_block + wm + mi * 16 + hi * 8;
            const float4 b0 = *(const float4*)(bias + m0);
            const float4 b1 = *(const float4*)(bias + m0 + 4);
            float bb[8] = {b0.x, b0.y, b0.z, b0.w, b1.x, b1.y, b1.z, b1.w};
#pragma unroll
            for (int ni = 0; ni < 4; ++ni) {
                int n = n_block + wn + ni * 16 + col;
                float vals[8];
#pragma unroll
                for (int i = 0; i < 8; ++i) vals[i] = acc[mi][ni][i] + bb[i];
                uint4 pk;
                pk.x = pack2(vals[0], vals[1]);
                pk.y = pack2(vals[2], vals[3]);
                pk.z = pack2(vals[4], vals[5]);
                pk.w = pack2(vals[6], vals[7]);
                *(uint4*)(outBf + ((size_t)b * NPIX + n) * CDIM + m0) = pk;
            }
        }
    } else {
        // transpose 16x16 f32 tile through LDS, then coalesced row stores
        __shared__ float tile[8][16][17];
        float g = *gammaPtr;
#pragma unroll
        for (int mi = 0; mi < 2; ++mi)
#pragma unroll
            for (int ni = 0; ni < 4; ++ni) {
#pragma unroll
                for (int i = 0; i < 8; ++i)
                    tile[wave][hi * 8 + i][col] = g * acc[mi][ni][i];
                __builtin_amdgcn_wave_barrier();
                int mrow = m_block + wm + mi * 16 + col;
                int nc   = n_block + wn + ni * 16 + hi * 8;
                size_t off = ((size_t)b * CDIM + mrow) * NPIX + nc;
                float o[8];
#pragma unroll
                for (int j = 0; j < 8; ++j)
                    o[j] = tile[wave][col][hi * 8 + j] + resid[off + j];
                float4 st0 = {o[0], o[1], o[2], o[3]};
                float4 st1 = {o[4], o[5], o[6], o[7]};
                *(float4*)(outF + off)     = st0;
                *(float4*)(outF + off + 4) = st1;
                __builtin_amdgcn_wave_barrier();
            }
    }
}

// ---------------------------------------------------------------------------
extern "C" void kernel_launch(void* const* d_in, const int* in_sizes, int n_in,
                              void* d_out, int out_size, void* d_ws, size_t ws_size,
                              hipStream_t stream) {
    const float* x    = (const float*)d_in[0];  // feature_map [B,C,H,W]
    const float* att  = (const float*)d_in[1];  // attention  [B,1,H,W]
    const float* Wq   = (const float*)d_in[2];  // [C,C]
    const float* bq   = (const float*)d_in[3];  // [C]
    const float* Wk   = (const float*)d_in[4];  // [C,1]
    const float* bk   = (const float*)d_in[5];  // [C]
    const float* Wv   = (const float*)d_in[6];  // [C,C]
    const float* bv   = (const float*)d_in[7];  // [C]
    const float* gam  = (const float*)d_in[8];  // [1]
    float* out = (float*)d_out;

    char* ws = (char*)d_ws;
    size_t o_xT  = 0;                                                    // B*N*C bf16
    size_t o_Wvb = o_xT  + (size_t)BDIM * NPIX * CDIM * 2;               // C*C bf16
    size_t o_vT  = o_Wvb + (size_t)CDIM * CDIM * 2;                      // B*N*C bf16
    size_t o_P   = o_vT  + (size_t)BDIM * NPIX * CDIM * 2;               // B*C*C bf16
    size_t o_xa  = o_P   + (size_t)BDIM * CDIM * CDIM * 2;               // B*C f32
    size_t o_xs  = o_xa  + (size_t)BDIM * CDIM * 4;
    size_t o_A   = o_xs  + (size_t)BDIM * CDIM * 4;
    size_t o_S   = o_A   + (size_t)BDIM * CDIM * 4;
    size_t o_sa  = o_S   + (size_t)BDIM * CDIM * 4;

    __bf16* xT  = (__bf16*)(ws + o_xT);
    __bf16* Wvb = (__bf16*)(ws + o_Wvb);
    __bf16* vT  = (__bf16*)(ws + o_vT);
    __bf16* P   = (__bf16*)(ws + o_P);
    float*  xa  = (float*)(ws + o_xa);
    float*  xs  = (float*)(ws + o_xs);
    float*  Ab  = (float*)(ws + o_A);
    float*  Sb  = (float*)(ws + o_S);
    float*  sa  = (float*)(ws + o_sa);

    // 1) reductions + bf16 transpose of x
    prep_kernel<<<BDIM * (CDIM / 8), 256, 0, stream>>>(x, att, xT, xa, xs);
    sa_kernel<<<BDIM, 256, 0, stream>>>(att, sa);
    cvt_kernel<<<(CDIM * CDIM / 4 + 255) / 256, 256, 0, stream>>>(Wv, Wvb, CDIM * CDIM);
    // 2) collapsed q-projection / score matvecs
    qsum_kernel<<<CDIM / 8, 256, 0, stream>>>(Wq, bq, xa, xs, sa, Ab, Sb);
    // 3) softmax -> P (bf16)
    softmax_kernel<<<BDIM * CDIM / 8, 256, 0, stream>>>(Wk, bk, Ab, Sb, P);
    // 4) v = Wv @ x + bv  (bf16 WMMA, transposed bf16 output)
    gemm_wmma_kernel<0><<<BDIM * 128, 256, 0, stream>>>(
        Wvb, (size_t)0, xT, bv, nullptr, nullptr, vT, nullptr);
    // 5) out = gamma * (P @ v) + x  (bf16 WMMA, fp32 output)
    gemm_wmma_kernel<1><<<BDIM * 128, 256, 0, stream>>>(
        P, (size_t)CDIM * CDIM, vT, nullptr, gam, x, nullptr, out);
}
